// GINLayer_2954937499914
// MI455X (gfx1250) — matrix operand order repaired and checked by reading, not compiled, not run on hardware
//
#include <hip/hip_runtime.h>
#include <stdint.h>

typedef __attribute__((ext_vector_type(16))) __bf16 v16bf;
typedef __attribute__((ext_vector_type(8)))  float  v8f;
typedef __attribute__((ext_vector_type(4)))  int    v4i;

#define DFEAT 512
#define LN_EPS 1e-5f

// ---- gfx1250 async global->LDS staging (ASYNCcnt path), with fallback -----
// Probe-revealed signature: param0 = AS(1) int4* (global src),
//                           param1 = AS(3) int4* (LDS dst), then imm offset, cpol.
#if defined(__AMDGCN__) && __has_builtin(__builtin_amdgcn_global_load_async_to_lds_b128)
#define HAVE_ASYNC_LDS 1
#define ASYNC_CP_B128(dst, src)                                              \
  __builtin_amdgcn_global_load_async_to_lds_b128(                            \
      (__attribute__((address_space(1))) v4i*)(src),                         \
      (__attribute__((address_space(3))) v4i*)(dst), 0, 0)
#if __has_builtin(__builtin_amdgcn_s_wait_asynccnt)
#define ASYNC_WAIT0() __builtin_amdgcn_s_wait_asynccnt(0)
#else
#define ASYNC_WAIT0() asm volatile("s_wait_asynccnt 0x0" ::: "memory")
#endif
#else
#define HAVE_ASYNC_LDS 0
#endif

// ---------------------------------------------------------------------------
// Pack W[o,k] (row-major 512x512) into WMMA-B fragment order, split into
// bf16 hi/lo pair (A*B = Ahi*Bhi + Alo*Bhi + Ahi*Blo gives ~fp32 accuracy).
// Layout: flat = ((kt*32 + nt)*32 + lane)*16 + elem
//   B element (k,n): kt=k/32, kk=k%32, nt=n/16, nn=n%16
//   lane = nn + 16*(kk>=16), elem = kk & 15
// ---------------------------------------------------------------------------
__global__ __launch_bounds__(256) void pack_w(const float* __restrict__ W,
                                              unsigned short* __restrict__ Whi,
                                              unsigned short* __restrict__ Wlo) {
  int flat = blockIdx.x * 256 + threadIdx.x;      // 0 .. 262143
  int elem = flat & 15;
  int lane = (flat >> 4) & 31;
  int nt   = (flat >> 9) & 31;
  int kt   = flat >> 14;
  int kk   = elem | ((lane >> 4) << 4);
  int o    = (nt << 4) + (lane & 15);
  int k    = (kt << 5) + kk;
  float v  = W[o * DFEAT + k];
  __bf16 h = (__bf16)v;
  __bf16 l = (__bf16)(v - (float)h);
  Whi[flat] = __builtin_bit_cast(unsigned short, h);
  Wlo[flat] = __builtin_bit_cast(unsigned short, l);
}

// h0 = (1+eps) * x   (float4 vectorized)
__global__ __launch_bounds__(256) void scale_x(const float* __restrict__ x,
                                               const float* __restrict__ eps,
                                               float* __restrict__ h0, int total4) {
  int i = blockIdx.x * 256 + threadIdx.x;
  if (i < total4) {
    float s = 1.0f + eps[0];
    float4 v = ((const float4*)x)[i];
    v.x *= s; v.y *= s; v.z *= s; v.w *= s;
    ((float4*)h0)[i] = v;
  }
}

// One wave32 per edge: gather x[src] (L2-resident), atomic-add into h0[dst].
__global__ __launch_bounds__(256) void edge_scatter(const float* __restrict__ x,
                                                    const int* __restrict__ ei,
                                                    float* __restrict__ h0, int E) {
  int w = threadIdx.x >> 5, lane = threadIdx.x & 31;
  int e = blockIdx.x * 8 + w;
  if (e >= E) return;
  int dst = ei[e];          // edge_index[0] = destination i
  int src = ei[E + e];      // edge_index[1] = source j
  const float4* xr = (const float4*)(x + (size_t)src * DFEAT);
  float* hr = h0 + (size_t)dst * DFEAT;
#pragma unroll
  for (int c = 0; c < 4; ++c) {
    float4 v = xr[lane + 32 * c];
    int col = (lane + 32 * c) * 4;
    __hip_atomic_fetch_add(hr + col + 0, v.x, __ATOMIC_RELAXED, __HIP_MEMORY_SCOPE_AGENT);
    __hip_atomic_fetch_add(hr + col + 1, v.y, __ATOMIC_RELAXED, __HIP_MEMORY_SCOPE_AGENT);
    __hip_atomic_fetch_add(hr + col + 2, v.z, __ATOMIC_RELAXED, __HIP_MEMORY_SCOPE_AGENT);
    __hip_atomic_fetch_add(hr + col + 3, v.w, __ATOMIC_RELAXED, __HIP_MEMORY_SCOPE_AGENT);
  }
}

// ---------------------------------------------------------------------------
// Fused GEMM: out[m,n] = epilogue( A[m,:] @ W^T[:,n] + bias[n] )
// Block tile: M=32 x N=512 (full row width -> LayerNorm fusable).
// 8 waves: wave(mw,nw) computes rows mw*16..+16, cols [n-tiles t*4+nw].
// K loop in 32-chunks; split-bf16 3-WMMA accumulation per tile.
// B staging uses GLOBAL_LOAD_ASYNC_TO_LDS_B128 (weights pre-swizzled to
// fragment order, so it is a straight 16B/lane copy).
// LN==0: ReLU epilogue. LN==1: LayerNorm epilogue.
// ---------------------------------------------------------------------------
template <int LN>
__global__ __launch_bounds__(256) void gin_gemm(
    const float* __restrict__ A, const unsigned short* __restrict__ Bhi_g,
    const unsigned short* __restrict__ Blo_g, const float* __restrict__ bias,
    const float* __restrict__ gamma, const float* __restrict__ beta,
    float* __restrict__ out, int N) {
  __shared__ __align__(32) unsigned short sBhi[16 * 512];  // 16 n-tiles, 16KB
  __shared__ __align__(32) unsigned short sBlo[16 * 512];
  __shared__ __align__(32) unsigned short sAhi[2 * 512];   // 2 m-tiles
  __shared__ __align__(32) unsigned short sAlo[2 * 512];
  __shared__ float redS[2][16][4];
  __shared__ float redQ[2][16][4];

  const int tid  = threadIdx.x;
  const int lane = tid & 31;
  const int w    = tid >> 5;
  const int mw   = w >> 2;  // 0..1
  const int nw   = w & 3;   // 0..3
  const int m0   = blockIdx.x * 32;

  v8f acc[8];
#pragma unroll
  for (int s = 0; s < 8; ++s)
#pragma unroll
    for (int r = 0; r < 8; ++r) acc[s][r] = 0.0f;

  v16bf ahi{}, alo{};

  for (int kt = 0; kt < 16; ++kt) {
#pragma unroll
    for (int nh = 0; nh < 2; ++nh) {
      __syncthreads();
      // ---- Stage B half: 16 n-tiles (16KB hi + 16KB lo), frag-order copy.
      {
        const uint4* srcH = (const uint4*)(Bhi_g + kt * 16384 + nh * 8192);
        const uint4* srcL = (const uint4*)(Blo_g + kt * 16384 + nh * 8192);
        uint4* dstH = (uint4*)sBhi;
        uint4* dstL = (uint4*)sBlo;
#if HAVE_ASYNC_LDS
#pragma unroll
        for (int i = 0; i < 4; ++i) {
          ASYNC_CP_B128(dstH + tid + 256 * i, srcH + tid + 256 * i);
          ASYNC_CP_B128(dstL + tid + 256 * i, srcL + tid + 256 * i);
        }
#else
#pragma unroll
        for (int i = 0; i < 4; ++i) {
          dstH[tid + 256 * i] = srcH[tid + 256 * i];
          dstL[tid + 256 * i] = srcL[tid + 256 * i];
        }
#endif
      }
      if (nh == 0) {
        // ---- Stage A chunk: 32 rows x 32 k, f32 -> split bf16 hi/lo.
        int mrow = tid >> 3;
        int kk4  = (tid & 7) << 2;
        int m    = m0 + mrow;
        float4 v = make_float4(0.f, 0.f, 0.f, 0.f);
        if (m < N) {
          v = *(const float4*)(A + (size_t)m * DFEAT + (kt << 5) + kk4);
          if (kt < 15)
            __builtin_prefetch(A + (size_t)m * DFEAT + ((kt + 1) << 5) + kk4, 0, 0);
        }
        // A frag map: lane = m%16 + 16*(kk>>4), elem = kk&15
        int lane_s = (mrow & 15) + ((kk4 >> 4) << 4);
        int eb     = kk4 & 15;
        int base   = ((mrow >> 4) << 9) + (lane_s << 4) + eb;
        float vf[4] = {v.x, v.y, v.z, v.w};
#pragma unroll
        for (int c = 0; c < 4; ++c) {
          __bf16 h = (__bf16)vf[c];
          __bf16 l = (__bf16)(vf[c] - (float)h);
          sAhi[base + c] = __builtin_bit_cast(unsigned short, h);
          sAlo[base + c] = __builtin_bit_cast(unsigned short, l);
        }
      }
#if HAVE_ASYNC_LDS
      ASYNC_WAIT0();   // async-LDS writes visible before the barrier
#endif
      __syncthreads();
      if (nh == 0) {
        ahi = *(const v16bf*)(const void*)(sAhi + (mw << 9) + (lane << 4));
        alo = *(const v16bf*)(const void*)(sAlo + (mw << 9) + (lane << 4));
      }
#pragma unroll
      for (int t = 0; t < 4; ++t) {
        int lt = (t << 2) + nw;
        v16bf bh = *(const v16bf*)(const void*)(sBhi + (lt << 9) + (lane << 4));
        v16bf bl = *(const v16bf*)(const void*)(sBlo + (lt << 9) + (lane << 4));
        int s = (nh << 2) + t;
        acc[s] = __builtin_amdgcn_wmma_f32_16x16x32_bf16(false, ahi, false, bh,
                                                         (short)0, acc[s], false, false);
        acc[s] = __builtin_amdgcn_wmma_f32_16x16x32_bf16(false, alo, false, bh,
                                                         (short)0, acc[s], false, false);
        acc[s] = __builtin_amdgcn_wmma_f32_16x16x32_bf16(false, ahi, false, bl,
                                                         (short)0, acc[s], false, false);
      }
    }
  }

  // ---- Epilogue. C layout: VGPR r, lanes0-15: (M=r, N=lane); lanes16-31: (M=8+r, N=lane-16)
  const int rowbase = m0 + (mw << 4) + ((lane >> 4) << 3);
  int ncol[8];
#pragma unroll
  for (int s = 0; s < 8; ++s) {
    int g = ((s >> 2) << 4) + ((s & 3) << 2) + nw;  // global n-tile
    ncol[s] = (g << 4) + (lane & 15);
  }

  if (LN == 0) {  // bias + ReLU
#pragma unroll
    for (int s = 0; s < 8; ++s) {
      float bv = bias[ncol[s]];
#pragma unroll
      for (int r = 0; r < 8; ++r) {
        int m = rowbase + r;
        if (m < N) out[(size_t)m * DFEAT + ncol[s]] = fmaxf(acc[s][r] + bv, 0.0f);
      }
    }
  } else {  // bias + LayerNorm over the 512-wide row owned by this block
    float vv[8][8];
#pragma unroll
    for (int s = 0; s < 8; ++s) {
      float bv = bias[ncol[s]];
#pragma unroll
      for (int r = 0; r < 8; ++r) vv[s][r] = acc[s][r] + bv;
    }
    float p[8], q[8];
#pragma unroll
    for (int r = 0; r < 8; ++r) {
      float ps = 0.f, qs = 0.f;
#pragma unroll
      for (int s = 0; s < 8; ++s) { ps += vv[s][r]; qs += vv[s][r] * vv[s][r]; }
      p[r] = ps; q[r] = qs;
    }
    // reduce across the 16 lanes of each half (masks stay inside the half)
#pragma unroll
    for (int msk = 1; msk <= 8; msk <<= 1) {
#pragma unroll
      for (int r = 0; r < 8; ++r) {
        p[r] += __shfl_xor(p[r], msk, 32);
        q[r] += __shfl_xor(q[r], msk, 32);
      }
    }
    if ((lane & 15) == 0) {
      int half = lane >> 4;
#pragma unroll
      for (int r = 0; r < 8; ++r) {
        redS[mw][(half << 3) + r][nw] = p[r];
        redQ[mw][(half << 3) + r][nw] = q[r];
      }
    }
    __syncthreads();
#pragma unroll
    for (int r = 0; r < 8; ++r) {
      int rowl = ((lane >> 4) << 3) + r;
      float sm = redS[mw][rowl][0] + redS[mw][rowl][1] + redS[mw][rowl][2] + redS[mw][rowl][3];
      float sq = redQ[mw][rowl][0] + redQ[mw][rowl][1] + redQ[mw][rowl][2] + redQ[mw][rowl][3];
      float mu  = sm * (1.0f / DFEAT);
      float var = sq * (1.0f / DFEAT) - mu * mu;
      float rs  = rsqrtf(var + LN_EPS);
      int m = rowbase + r;
      if (m < N) {
#pragma unroll
        for (int s = 0; s < 8; ++s) {
          int n = ncol[s];
          out[(size_t)m * DFEAT + n] = (vv[s][r] - mu) * rs * gamma[n] + beta[n];
        }
      }
    }
  }
}

extern "C" void kernel_launch(void* const* d_in, const int* in_sizes, int n_in,
                              void* d_out, int out_size, void* d_ws, size_t ws_size,
                              hipStream_t stream) {
  const float* x     = (const float*)d_in[0];
  const int*   ei    = (const int*)d_in[1];
  const float* eps   = (const float*)d_in[2];
  const float* W1    = (const float*)d_in[3];
  const float* b1    = (const float*)d_in[4];
  const float* W2    = (const float*)d_in[5];
  const float* b2    = (const float*)d_in[6];
  const float* gamma = (const float*)d_in[7];
  const float* beta  = (const float*)d_in[8];

  const int N = in_sizes[0] / DFEAT;
  const int E = in_sizes[1] / 2;

  // ws layout: 4 x 512KB packed weights, then h0 (N*512 f32)
  unsigned short* W1hi = (unsigned short*)d_ws;
  unsigned short* W1lo = W1hi + 262144;
  unsigned short* W2hi = W1lo + 262144;
  unsigned short* W2lo = W2hi + 262144;
  float* h0 = (float*)((char*)d_ws + (size_t)4 * 262144 * 2);
  float* h1 = (float*)d_out;  // safe: each GEMM-2 block reads its rows before writing them

  pack_w<<<1024, 256, 0, stream>>>(W1, W1hi, W1lo);
  pack_w<<<1024, 256, 0, stream>>>(W2, W2hi, W2lo);

  int total4 = N * (DFEAT / 4);
  scale_x<<<(total4 + 255) / 256, 256, 0, stream>>>(x, eps, h0, total4);
  edge_scatter<<<(E + 7) / 8, 256, 0, stream>>>(x, ei, h0, E);

  int gblocks = (N + 31) / 32;
  gin_gemm<0><<<gblocks, 256, 0, stream>>>(h0, W1hi, W1lo, b1, nullptr, nullptr, h1, N);
  gin_gemm<1><<<gblocks, 256, 0, stream>>>(h1, W2hi, W2lo, b2, gamma, beta, (float*)d_out, N);
}
